// ContextualLoss_66176856097431
// MI455X (gfx1250) — compile-verified, matrix operand order checked by reading, not run on hardware
//
#include <hip/hip_runtime.h>
#include <hip/hip_bf16.h>
#include <math.h>

typedef _Float16 half8  __attribute__((ext_vector_type(8)));
typedef _Float16 half16 __attribute__((ext_vector_type(16)));
typedef float    f32x8  __attribute__((ext_vector_type(8)));

constexpr int kN  = 2;
constexpr int kC  = 256;
constexpr int kHW = 9216;     // 96*96
constexpr int kBT = kHW / 64; // 144 blocks of 64 rows/cols per batch

// ---- order-preserving float<->uint maps (for atomic min/max over floats) ----
__device__ __forceinline__ unsigned encF(float f) {
    unsigned u = __float_as_uint(f);
    return (u & 0x80000000u) ? ~u : (u | 0x80000000u);
}
__device__ __forceinline__ float decF(unsigned u) {
    u = (u & 0x80000000u) ? (u & 0x7FFFFFFFu) : ~u;
    return __uint_as_float(u);
}

// ---- init per-row / per-col stats (ws is poisoned by harness each run) ----
__global__ void init_stats(unsigned* __restrict__ rowmin, float* __restrict__ rowsum,
                           unsigned* __restrict__ colmax) {
    int i = blockIdx.x * 256 + threadIdx.x;
    if (i < kN * kHW) { rowmin[i] = 0xFFFFFFFFu; rowsum[i] = 0.f; colmax[i] = 0u; }
}

// ---- per-channel mean of y over (n,h,w) ----
__global__ void channel_mean(const float* __restrict__ y, float* __restrict__ mu) {
    const int c = blockIdx.x;
    __shared__ float sh[256];
    float s = 0.f;
    for (int i = threadIdx.x; i < kN * kHW; i += 256) {
        int n = i / kHW, p = i % kHW;
        s += y[((size_t)(n * kC + c)) * kHW + p];
    }
    sh[threadIdx.x] = s; __syncthreads();
    for (int off = 128; off >= 1; off >>= 1) {
        if (threadIdx.x < off) sh[threadIdx.x] += sh[threadIdx.x + off];
        __syncthreads();
    }
    if (threadIdx.x == 0) mu[c] = sh[0] / (float)(kN * kHW);
}

// ---- center + channel-L2-normalize; write f16 (n,p,c) row-major ----
__global__ void normalize_kernel(const float* __restrict__ x, const float* __restrict__ y,
                                 const float* __restrict__ mu,
                                 _Float16* __restrict__ Xh, _Float16* __restrict__ Yh) {
    const int np = blockIdx.x;
    const int n = np / kHW, p = np % kHW;
    const int c = threadIdx.x;
    __shared__ float shx[256], shy[256];
    const size_t idx = ((size_t)(n * kC + c)) * kHW + p;
    const float m  = mu[c];
    const float xv = x[idx] - m;
    const float yv = y[idx] - m;
    shx[c] = xv * xv; shy[c] = yv * yv;
    __syncthreads();
    for (int off = 128; off >= 1; off >>= 1) {
        if (c < off) { shx[c] += shx[c + off]; shy[c] += shy[c + off]; }
        __syncthreads();
    }
    const float nx = fmaxf(sqrtf(shx[0]), 1e-12f);
    const float ny = fmaxf(sqrtf(shy[0]), 1e-12f);
    const size_t o = ((size_t)n * kHW + p) * kC + c;
    Xh[o] = (_Float16)(xv / nx);
    Yh[o] = (_Float16)(yv / ny);
}

// ---- load one 16x32 f16 tile in the CDNA5 A-matrix VGPR layout ----
// lane 0..15 : row M=lane,    K = {kb..kb+7, kb+16..kb+23} with kb=0
// lane 16..31: row M=lane-16, K chunks with kb=8
__device__ __forceinline__ half16 load_tile16x32(const _Float16* __restrict__ base, int lane) {
    const int row = lane & 15;
    const int kb  = (lane >> 4) << 3;
    const _Float16* p = base + (size_t)row * kC + kb;
    half8 lo = *(const half8*)(p);
    half8 hi = *(const half8*)(p + 16);
    return __builtin_shufflevector(lo, hi, 0,1,2,3,4,5,6,7,8,9,10,11,12,13,14,15);
}

// ---- fused GEMM (dist = 1 - Xn·Ynᵀ) with pass-dependent epilogue ----
// PASS 1: row-min of dist          -> rowmin (encoded uint atomicMin)
// PASS 2: row-sum of exp(...)      -> rowsum (f32 atomicAdd)
// PASS 3: col-max of w/rowsum      -> colmax (encoded uint atomicMax)
template <int PASS>
__global__ void __launch_bounds__(256)
cx_gemm(const _Float16* __restrict__ Xh, const _Float16* __restrict__ Yh,
        unsigned* __restrict__ rowmin, float* __restrict__ rowsum,
        unsigned* __restrict__ colmax) {
    const int lane = threadIdx.x & 31;
    const int wid  = blockIdx.x * (blockDim.x >> 5) + (threadIdx.x >> 5);
    if (wid >= kN * kBT * kBT) return;
    const int n    = wid / (kBT * kBT);
    const int rem  = wid % (kBT * kBT);
    const int bi   = rem / kBT;       // 64-row block (queries)
    const int bj   = rem % kBT;       // 64-col block (keys)
    const int half = lane >> 4;

    const _Float16* Ab = Xh + ((size_t)n * kHW + (size_t)bi * 64) * kC;
    const _Float16* Bb = Yh + ((size_t)n * kHW + (size_t)bj * 64) * kC;

    f32x8 acc[4][4] = {};

    // K = 256, step 32: 8 iterations, 16 WMMA each. Operands are L2-resident.
    for (int k0 = 0; k0 < kC; k0 += 32) {
        half16 a[4], b[4];
        #pragma unroll
        for (int i = 0; i < 4; ++i) a[i] = load_tile16x32(Ab + (size_t)i * 16 * kC + k0, lane);
        #pragma unroll
        for (int j = 0; j < 4; ++j) b[j] = load_tile16x32(Bb + (size_t)j * 16 * kC + k0, lane);
        #pragma unroll
        for (int i = 0; i < 4; ++i)
            #pragma unroll
            for (int j = 0; j < 4; ++j)
                acc[i][j] = __builtin_amdgcn_wmma_f32_16x16x32_f16(
                    false, a[i], false, b[j], (short)0, acc[i][j], false, false);
    }

    // C/D layout: VGPR r holds (M = r + 8*half, N = lane&15) of each 16x16 tile.
    if (PASS == 1) {
        #pragma unroll
        for (int i = 0; i < 4; ++i) {
            #pragma unroll
            for (int r = 0; r < 8; ++r) {
                float m = 3.0e38f;
                #pragma unroll
                for (int j = 0; j < 4; ++j) m = fminf(m, 1.0f - acc[i][j][r]);
                #pragma unroll
                for (int off = 8; off >= 1; off >>= 1) m = fminf(m, __shfl_xor(m, off, 16));
                if ((lane & 15) == 0) {
                    int p = bi * 64 + i * 16 + 8 * half + r;
                    atomicMin(&rowmin[n * kHW + p], encF(m));
                }
            }
        }
    } else if (PASS == 2) {
        #pragma unroll
        for (int i = 0; i < 4; ++i) {
            #pragma unroll
            for (int r = 0; r < 8; ++r) {
                const int p = bi * 64 + i * 16 + 8 * half + r;
                const float rmv = decF(rowmin[n * kHW + p]) + 1e-5f;
                float s = 0.f;
                #pragma unroll
                for (int j = 0; j < 4; ++j) {
                    float d = 1.0f - acc[i][j][r];
                    s += __expf((1.0f - d / rmv) * 2.0f);   // /BAND_WIDTH(0.5)
                }
                #pragma unroll
                for (int off = 8; off >= 1; off >>= 1) s += __shfl_xor(s, off, 16);
                if ((lane & 15) == 0) atomicAdd(&rowsum[n * kHW + p], s);
            }
        }
    } else {
        float cmax[4] = {0.f, 0.f, 0.f, 0.f};
        #pragma unroll
        for (int i = 0; i < 4; ++i) {
            #pragma unroll
            for (int r = 0; r < 8; ++r) {
                const int p = bi * 64 + i * 16 + 8 * half + r;
                const float rmv    = decF(rowmin[n * kHW + p]) + 1e-5f;
                const float inv_rs = 1.0f / rowsum[n * kHW + p];
                #pragma unroll
                for (int j = 0; j < 4; ++j) {
                    float d = 1.0f - acc[i][j][r];
                    float w = __expf((1.0f - d / rmv) * 2.0f);
                    cmax[j] = fmaxf(cmax[j], w * inv_rs);
                }
            }
        }
        #pragma unroll
        for (int j = 0; j < 4; ++j) {
            float cm = fmaxf(cmax[j], __shfl_xor(cmax[j], 16, 32)); // merge M / M+8 halves
            if (lane < 16) {
                int q = bj * 64 + j * 16 + lane;
                atomicMax(&colmax[n * kHW + q], encF(cm));
            }
        }
    }
}

// ---- final: loss = mean_n( -log( mean_q(colmax) + eps ) ) ----
__global__ void finalize(const unsigned* __restrict__ colmax, float* __restrict__ out) {
    __shared__ float sh[256];
    float loss = 0.f;
    for (int n = 0; n < kN; ++n) {
        float s = 0.f;
        for (int q = threadIdx.x; q < kHW; q += 256) s += decF(colmax[n * kHW + q]);
        sh[threadIdx.x] = s; __syncthreads();
        for (int off = 128; off >= 1; off >>= 1) {
            if (threadIdx.x < off) sh[threadIdx.x] += sh[threadIdx.x + off];
            __syncthreads();
        }
        float cxn = sh[0] / (float)kHW;
        loss += -logf(cxn + 1e-5f);
        __syncthreads();
    }
    if (threadIdx.x == 0) out[0] = loss / (float)kN;
}

extern "C" void kernel_launch(void* const* d_in, const int* in_sizes, int n_in,
                              void* d_out, int out_size, void* d_ws, size_t ws_size,
                              hipStream_t stream) {
    (void)in_sizes; (void)n_in; (void)out_size; (void)ws_size;
    const float* x = (const float*)d_in[0];
    const float* y = (const float*)d_in[1];
    float* out = (float*)d_out;

    // workspace layout (~19.2 MB total)
    char* ws = (char*)d_ws;
    float*    mu     = (float*)ws;                                  // 1 KB
    _Float16* Xh     = (_Float16*)(ws + 1024);                      // 9.44 MB
    _Float16* Yh     = Xh + (size_t)kN * kHW * kC;                  // 9.44 MB
    unsigned* rowmin = (unsigned*)(Yh + (size_t)kN * kHW * kC);     // 72 KB
    float*    rowsum = (float*)(rowmin + kN * kHW);                 // 72 KB
    unsigned* colmax = (unsigned*)(rowsum + kN * kHW);              // 72 KB

    init_stats<<<(kN * kHW + 255) / 256, 256, 0, stream>>>(rowmin, rowsum, colmax);
    channel_mean<<<kC, 256, 0, stream>>>(y, mu);
    normalize_kernel<<<kN * kHW, 256, 0, stream>>>(x, y, mu, Xh, Yh);

    const int waves  = kN * kBT * kBT;       // 41472 waves, 64x64 output each
    const int blocks = (waves + 7) / 8;      // 8 waves per 256-thread block
    cx_gemm<1><<<blocks, 256, 0, stream>>>(Xh, Yh, rowmin, rowsum, colmax);
    cx_gemm<2><<<blocks, 256, 0, stream>>>(Xh, Yh, rowmin, rowsum, colmax);
    cx_gemm<3><<<blocks, 256, 0, stream>>>(Xh, Yh, rowmin, rowsum, colmax);

    finalize<<<1, 256, 0, stream>>>(colmax, out);
}
